// PredicateVisual_acc_28226525069939
// MI455X (gfx1250) — matrix operand order repaired and checked by reading, not compiled, not run on hardware
//
#include <hip/hip_runtime.h>
#include <hip/hip_bf16.h>

typedef __bf16 bf16;
typedef __attribute__((ext_vector_type(16))) __bf16 v16bf;
typedef __attribute__((ext_vector_type(8)))  float  v8f;

#if __has_builtin(__builtin_amdgcn_global_load_async_to_lds_b128)
#define USE_ASYNC_LDS 1
#else
#define USE_ASYNC_LDS 0
#endif

#if __has_builtin(__builtin_amdgcn_s_wait_asynccnt)
#define WAIT_ASYNC(n) __builtin_amdgcn_s_wait_asynccnt(n)
#else
#define WAIT_ASYNC(n) asm volatile("s_wait_asynccnt %0" ::"i"(n) : "memory")
#endif

#if USE_ASYNC_LDS
// builtin signature (from hipcc diagnostic): params are pointers to
// 'int __attribute__((vector_size(16)))' in global / shared address spaces.
typedef int v4i_g __attribute__((vector_size(16)));
typedef __attribute__((address_space(1))) v4i_g* gas_b128_t;
typedef __attribute__((address_space(3))) v4i_g* las_b128_t;

__device__ __forceinline__ void async_copy_b128(const void* g, void* l)
{
    // flat->AS1: same 64-bit value. flat LDS ptr low 32 bits == LDS offset (ISA 10.2).
    __builtin_amdgcn_global_load_async_to_lds_b128(
        (gas_b128_t)(uintptr_t)g,
        (las_b128_t)(uint32_t)(uintptr_t)l,
        0, 0);
}
#endif

// ---------------- problem constants ----------------
constexpr int Bsz  = 1024;      // batch
constexpr int VD   = 12288;     // visual_d*3
constexpr int OBJ  = 4096;      // obj visual_d
constexpr int H1   = 2048;      // hidden of forward3 #1
constexpr int H2   = 4096;      // hidden of forward3 #2
constexpr int ED   = 600;      // embedding dim
constexpr int EDp  = 640;       // padded N for 600-wide layers
constexpr int ZREAL= 1800;      // concat width
constexpr int ZK   = 1856;      // z row stride / K of fusion hidden (29*64)
constexpr int HN   = 1920;      // fusion hidden padded N / K of final (15*128, 30*64)
constexpr int NNEG = 128;
constexpr int NCAND= NNEG + 1;

// ---------------- GEMM tiling ----------------
constexpr int BM = 128, BN = 128, BK = 64;
constexpr int LDK = BK + 8;     // 72 bf16 -> 144B rows, 16B aligned, bank-spread

union FragAB { uint4 q[2]; v16bf v; };

// C[m, colOff+n] = act( A[m,:] . B[n,:] + bias[n] )   (B row-major [Npad,K])
template<bool RELU, bool OUT_F32>
__global__ __launch_bounds__(256, 1)
void gemm_bf16_wmma(const bf16* __restrict__ A, int lda,
                    const bf16* __restrict__ B,
                    const float* __restrict__ bias, int nBias,
                    void* __restrict__ C, int ldc, int colOff, int nStore,
                    int K)
{
    __shared__ __align__(16) bf16 As[2][BM * LDK];
    __shared__ __align__(16) bf16 Bs[2][BN * LDK];

    const int tid  = threadIdx.x;
    const int lane = tid & 31;
    const int wave = tid >> 5;
    const int wm   = wave >> 2;           // 0..1  (64 rows each)
    const int wn   = wave & 3;            // 0..3  (32 cols each)
    const int tileM = blockIdx.y * BM;
    const int tileN = blockIdx.x * BN;

    // cooperative staging: 256 thr * 64B = 16KB per matrix per K-tile
    const int ldRow = tid >> 1;           // 0..127
    const int ldCol = (tid & 1) * 32;     // bf16 column within tile
    const bf16* gA = A + (size_t)(tileM + ldRow) * lda + ldCol;
    const bf16* gB = B + (size_t)(tileN + ldRow) * K   + ldCol;

#if USE_ASYNC_LDS
    // async DMA: global -> LDS directly, tracked by ASYNCcnt (8 x b128 per thread)
    auto aload = [&](int buf, int kt) {
        const bf16* ga = gA + (size_t)kt * BK;
        const bf16* gb = gB + (size_t)kt * BK;
        bf16* sa = &As[buf][ldRow * LDK + ldCol];
        bf16* sb = &Bs[buf][ldRow * LDK + ldCol];
#pragma unroll
        for (int i = 0; i < 4; ++i)
            async_copy_b128(ga + i * 8, sa + i * 8);
#pragma unroll
        for (int i = 0; i < 4; ++i)
            async_copy_b128(gb + i * 8, sb + i * 8);
    };
#else
    uint4 ra[4], rb[4];
    auto gload = [&](int kt) {
        const uint4* pa = (const uint4*)(gA + (size_t)kt * BK);
        const uint4* pb = (const uint4*)(gB + (size_t)kt * BK);
#pragma unroll
        for (int i = 0; i < 4; ++i) { ra[i] = pa[i]; rb[i] = pb[i]; }
    };
    auto sstore = [&](int buf) {
        uint4* sa = (uint4*)&As[buf][ldRow * LDK + ldCol];
        uint4* sb = (uint4*)&Bs[buf][ldRow * LDK + ldCol];
#pragma unroll
        for (int i = 0; i < 4; ++i) { sa[i] = ra[i]; sb[i] = rb[i]; }
    };
#endif

    v8f acc[4][2];
#pragma unroll
    for (int mi = 0; mi < 4; ++mi)
#pragma unroll
        for (int ni = 0; ni < 2; ++ni)
#pragma unroll
            for (int r = 0; r < 8; ++r) acc[mi][ni][r] = 0.0f;

    // WMMA 16-bit fragment lane mapping (wave32)
    const int aM = lane & 15;             // A: row within 16
    const int aK = (lane >> 4) * 8;       // A: two 8-elem K chunks at aK, aK+16
    const int bN = lane & 15;             // B: col within 16
    const int bK = (lane >> 4) * 8;       // B: two 8-elem K chunks at bK, bK+16

    auto compute = [&](int buf) {
#pragma unroll
        for (int kk = 0; kk < BK; kk += 32) {
            FragAB a[4], b[2];
#pragma unroll
            for (int mi = 0; mi < 4; ++mi) {
                const bf16* p = &As[buf][(wm * 64 + mi * 16 + aM) * LDK + kk + aK];
                a[mi].q[0] = *(const uint4*)p;
                a[mi].q[1] = *(const uint4*)(p + 16);
            }
#pragma unroll
            for (int ni = 0; ni < 2; ++ni) {
                const bf16* p = &Bs[buf][(wn * 32 + ni * 16 + bN) * LDK + kk + bK];
                b[ni].q[0] = *(const uint4*)p;
                b[ni].q[1] = *(const uint4*)(p + 16);
            }
#pragma unroll
            for (int mi = 0; mi < 4; ++mi)
#pragma unroll
                for (int ni = 0; ni < 2; ++ni)
                    acc[mi][ni] = __builtin_amdgcn_wmma_f32_16x16x32_bf16(
                        false, a[mi].v, false, b[ni].v,
                        (short)0, acc[mi][ni], false, false);
        }
    };

    const int KT = K / BK;
#if USE_ASYNC_LDS
    aload(0, 0);
    for (int kt = 0; kt < KT; ++kt) {
        __syncthreads();                       // everyone done reading buf[(kt+1)&1]
        if (kt + 1 < KT) {
            aload((kt + 1) & 1, kt + 1);       // 8 new in flight
            WAIT_ASYNC(8);                     // tile kt's 8 (older, in-order) landed
        } else {
            WAIT_ASYNC(0);
        }
        __syncthreads();                       // cross-wave visibility of buf[kt&1]
        compute(kt & 1);
    }
#else
    gload(0);
    sstore(0);
    for (int kt = 0; kt < KT; ++kt) {
        __syncthreads();
        if (kt + 1 < KT) gload(kt + 1);
        compute(kt & 1);
        if (kt + 1 < KT) sstore((kt + 1) & 1);
    }
#endif

    // epilogue: bias + activation + store (C/D layout: lane l -> n=l&15, m=r+8*(l>>4))
#pragma unroll
    for (int ni = 0; ni < 2; ++ni) {
        const int n = tileN + wn * 32 + ni * 16 + (lane & 15);
        const float bv = (n < nBias) ? bias[n] : 0.0f;
        if (n < nStore) {
#pragma unroll
            for (int mi = 0; mi < 4; ++mi) {
#pragma unroll
                for (int r = 0; r < 8; ++r) {
                    const int m = tileM + wm * 64 + mi * 16 + (lane >> 4) * 8 + r;
                    float v = acc[mi][ni][r] + bv;
                    if (RELU) v = fmaxf(v, 0.0f);
                    if (OUT_F32)
                        ((float*)C)[(size_t)m * ldc + colOff + n] = v;
                    else
                        ((bf16*)C)[(size_t)m * ldc + colOff + n] = (bf16)v;
                }
            }
        }
    }
}

// relu(x) then f32 -> bf16, vectorized x4
__global__ void relu_cvt_kernel(const float* __restrict__ x, bf16* __restrict__ y, long long n4)
{
    long long i = (long long)blockIdx.x * blockDim.x + threadIdx.x;
    const long long stride = (long long)gridDim.x * blockDim.x;
    for (; i < n4; i += stride) {
        float4 v = ((const float4*)x)[i];
        bf16* o = y + i * 4;
        o[0] = (bf16)fmaxf(v.x, 0.0f);
        o[1] = (bf16)fmaxf(v.y, 0.0f);
        o[2] = (bf16)fmaxf(v.z, 0.0f);
        o[3] = (bf16)fmaxf(v.w, 0.0f);
    }
}

// weight [N,K] f32 -> [Npad,Kpad] bf16 with zero padding
__global__ void cvt_pad_kernel(const float* __restrict__ w, bf16* __restrict__ o,
                               int N, int K, int Npad, int Kpad)
{
    long long total = (long long)Npad * Kpad;
    long long i = (long long)blockIdx.x * blockDim.x + threadIdx.x;
    const long long stride = (long long)gridDim.x * blockDim.x;
    for (; i < total; i += stride) {
        int r = (int)(i / Kpad), c = (int)(i % Kpad);
        float v = (r < N && c < K) ? w[(size_t)r * K + c] : 0.0f;
        o[i] = (bf16)v;
    }
}

// one wave per (batch, candidate): -||relu(label - emb)||_2
__global__ void score_kernel(const float* __restrict__ labels,
                             const float* __restrict__ emb,
                             const int* __restrict__ pls,
                             const int* __restrict__ nls,
                             float* __restrict__ out)
{
    const int gw   = (int)(((long long)blockIdx.x * blockDim.x + threadIdx.x) >> 5);
    const int lane = threadIdx.x & 31;
    if (gw >= Bsz * NCAND) return;
    const int b = gw / NCAND, c = gw % NCAND;
    const int row = (c == 0) ? pls[b] : nls[b * NNEG + (c - 1)];
    const float* L = labels + (size_t)row * ED;
    const float* E = emb    + (size_t)b   * ED;
    float s = 0.0f;
    for (int j = lane; j < ED; j += 32) {
        float d = fmaxf(L[j] - E[j], 0.0f);
        s += d * d;
    }
#pragma unroll
    for (int off = 16; off > 0; off >>= 1) s += __shfl_xor(s, off, 32);
    if (lane == 0) out[gw] = -sqrtf(s);
}

static inline size_t al256(size_t x) { return (x + 255) & ~(size_t)255; }

extern "C" void kernel_launch(void* const* d_in, const int* in_sizes, int n_in,
                              void* d_out, int out_size, void* d_ws, size_t ws_size,
                              hipStream_t stream)
{
    const float* vfs    = (const float*)d_in[0];
    const float* labels = (const float*)d_in[1];
    const int*   pls    = (const int*)  d_in[2];
    const int*   nls    = (const int*)  d_in[3];
    const float* W_h1 = (const float*)d_in[4];  const float* b_h1 = (const float*)d_in[5];
    const float* W_e1 = (const float*)d_in[6];  const float* b_e1 = (const float*)d_in[7];
    const float* W_h2 = (const float*)d_in[8];  const float* b_h2 = (const float*)d_in[9];
    const float* W_e2 = (const float*)d_in[10]; const float* b_e2 = (const float*)d_in[11];
    const float* W_h  = (const float*)d_in[12]; const float* b_h  = (const float*)d_in[13];
    const float* W_e  = (const float*)d_in[14]; const float* b_e  = (const float*)d_in[15];
    float* out = (float*)d_out;

    // ---- workspace carve-up (bf16 activations/weights, f32 emb) ----
    char* ws = (char*)d_ws;
    size_t off = 0;
    auto take = [&](size_t bytes) { char* p = ws + off; off = al256(off + bytes); return p; };
    bf16* Xbf  = (bf16*)take((size_t)Bsz * VD * 2);
    bf16* wh1  = (bf16*)take((size_t)H1 * OBJ * 2);
    bf16* we1  = (bf16*)take((size_t)EDp * H1 * 2);
    bf16* wh2  = (bf16*)take((size_t)H2 * VD * 2);
    bf16* we2  = (bf16*)take((size_t)EDp * H2 * 2);
    bf16* whf  = (bf16*)take((size_t)HN * ZK * 2);
    bf16* wef  = (bf16*)take((size_t)EDp * HN * 2);
    bf16* h1s  = (bf16*)take((size_t)Bsz * H1 * 2);
    bf16* h1o  = (bf16*)take((size_t)Bsz * H1 * 2);
    bf16* h2   = (bf16*)take((size_t)Bsz * H2 * 2);
    bf16* z    = (bf16*)take((size_t)Bsz * ZK * 2);
    bf16* hf   = (bf16*)take((size_t)Bsz * HN * 2);
    float* emb = (float*)take((size_t)Bsz * ED * 4);
    (void)ws_size; (void)n_in; (void)in_sizes; (void)out_size;

    // ---- stage 0: conversions (relu fused where reference applies it pre-GEMM) ----
    relu_cvt_kernel<<<4096, 256, 0, stream>>>(vfs, Xbf, (long long)Bsz * VD / 4);
    cvt_pad_kernel<<<2048, 256, 0, stream>>>(W_h1, wh1, H1, OBJ, H1, OBJ);
    cvt_pad_kernel<<<1024, 256, 0, stream>>>(W_e1, we1, ED, H1, EDp, H1);
    cvt_pad_kernel<<<8192, 256, 0, stream>>>(W_h2, wh2, H2, VD, H2, VD);
    cvt_pad_kernel<<<1024, 256, 0, stream>>>(W_e2, we2, ED, H2, EDp, H2);
    cvt_pad_kernel<<<1024, 256, 0, stream>>>(W_h,  whf, ZREAL, ZREAL, HN, ZK);
    cvt_pad_kernel<<<1024, 256, 0, stream>>>(W_e,  wef, ED, ZREAL, EDp, HN);

    const dim3 blk(256);
    // ---- forward3 #1 on sbj / obj slices, #2 on full vfs ----
    gemm_bf16_wmma<true, false><<<dim3(H1 / BN, Bsz / BM), blk, 0, stream>>>(
        Xbf, VD, wh1, b_h1, H1, h1s, H1, 0, H1, OBJ);                 // h1_sbj
    gemm_bf16_wmma<true, false><<<dim3(H1 / BN, Bsz / BM), blk, 0, stream>>>(
        Xbf + (VD - OBJ), VD, wh1, b_h1, H1, h1o, H1, 0, H1, OBJ);    // h1_obj
    gemm_bf16_wmma<true, false><<<dim3(H2 / BN, Bsz / BM), blk, 0, stream>>>(
        Xbf, VD, wh2, b_h2, H2, h2, H2, 0, H2, VD);                   // h2_pre

    // ---- embed layers, fused with the z=relu(concat) into slabs of z ----
    gemm_bf16_wmma<true, false><<<dim3(EDp / BN, Bsz / BM), blk, 0, stream>>>(
        h1s, H1, we1, b_e1, ED, z, ZK, 0,        ED, H1);             // z[:,0:600]
    gemm_bf16_wmma<true, false><<<dim3(EDp / BN, Bsz / BM), blk, 0, stream>>>(
        h2,  H2, we2, b_e2, ED, z, ZK, ED,       ED, H2);             // z[:,600:1200]
    gemm_bf16_wmma<true, false><<<dim3(EDp / BN, Bsz / BM), blk, 0, stream>>>(
        h1o, H1, we1, b_e1, ED, z, ZK, 2 * ED,   ED, H1);             // z[:,1200:1800]

    // ---- fusion: h = relu(z W_h^T + b_h) (stores zeroed pads), emb = h W_e^T + b_e ----
    gemm_bf16_wmma<true, false><<<dim3(HN / BN, Bsz / BM), blk, 0, stream>>>(
        z, ZK, whf, b_h, ZREAL, hf, HN, 0, HN, ZK);
    gemm_bf16_wmma<false, true><<<dim3(EDp / BN, Bsz / BM), blk, 0, stream>>>(
        hf, HN, wef, b_e, ED, emb, ED, 0, ED, HN);

    // ---- partial-order similarity scores ----
    const int totalWaves = Bsz * NCAND;                // 132096
    const int blocks = (totalWaves * 32 + 255) / 256;  // 8 waves / block
    score_kernel<<<blocks, 256, 0, stream>>>(labels, emb, pls, nls, out);
}
